// AudioAdapterAttnProcessor_44444321579513
// MI455X (gfx1250) — matrix operand order repaired and checked by reading, not compiled, not run on hardware
//
#include <hip/hip_runtime.h>
#include <hip/hip_bf16.h>

typedef __attribute__((ext_vector_type(16))) _Float16 v16h;
typedef __attribute__((ext_vector_type(8)))  float    v8f;

#define WMMA_F16(a, b, c) \
  __builtin_amdgcn_wmma_f32_16x16x32_f16(false, (a), false, (b), (short)0, (c), false, false)

static constexpr int Bsz  = 4;
static constexpr int S    = 4096;
static constexpr int D    = 1024;
static constexpr int Ctx  = 768;
static constexpr int H    = 16;
static constexpr int HD   = 64;
static constexpr int STXT = 77;
static constexpr int SAUD = 16;
static constexpr int STP  = 96;   // text kv padded to 6 tiles of 16
static constexpr int SAP  = 32;   // audio kv padded to one WMMA k-step

// ---- 16-lane butterfly reduction via ds_swizzle (group-of-32, xor mode) ----
template <int XM>
__device__ __forceinline__ float swz_xor(float x) {
  int v = __builtin_amdgcn_ds_swizzle(__float_as_int(x), (XM << 10) | 0x1f);
  return __int_as_float(v);
}

// ---- async global->LDS copy (CDNA5, tracked by ASYNCcnt) ----
__device__ __forceinline__ void async_copy_b128(uint32_t lds_off, const void* gptr) {
  asm volatile("global_load_async_to_lds_b128 %0, %1, off"
               :: "v"(lds_off), "v"(gptr) : "memory");
}
__device__ __forceinline__ void wait_async0() {
  asm volatile("s_wait_asynccnt 0x0" ::: "memory");
}

// ---------------------------------------------------------------------------
// fp32 -> fp16 convert (grid-stride)
// ---------------------------------------------------------------------------
__global__ __launch_bounds__(256) void cvt_f32_f16(const float* __restrict__ x,
                                                   _Float16* __restrict__ y, int n) {
  int i = blockIdx.x * 256 + threadIdx.x;
  int stride = gridDim.x * 256;
  for (; i < n; i += stride) y[i] = (_Float16)x[i];
}

// ---------------------------------------------------------------------------
// fp32 [K][N] -> fp16 transposed [N][K]
// ---------------------------------------------------------------------------
__global__ __launch_bounds__(256) void cvt_transpose_f32_f16(
    const float* __restrict__ W, _Float16* __restrict__ Wt, int K, int N) {
  int idx = blockIdx.x * 256 + threadIdx.x;
  if (idx < K * N) {
    int n = idx % N;
    int k = idx / N;
    Wt[(size_t)n * K + k] = (_Float16)W[idx];
  }
}

// ---------------------------------------------------------------------------
// KV projections (small GEMMs, scalar f32 math), write padded f16 layouts:
//   Kt/Vt: [b][h][96][64], Ka/Va: [b][h][32][64]; padded tokens = 0
// ---------------------------------------------------------------------------
__global__ __launch_bounds__(256) void kv_proj_kernel(
    const float* __restrict__ enc, const float* __restrict__ aud,
    const float* __restrict__ Wk, const float* __restrict__ Wv,
    const float* __restrict__ Wka, const float* __restrict__ Wva,
    _Float16* __restrict__ Kt, _Float16* __restrict__ Vt,
    _Float16* __restrict__ Ka, _Float16* __restrict__ Va) {
  size_t idx = (size_t)blockIdx.x * 256 + threadIdx.x;  // Bsz*128*1024 total
  int d  = (int)(idx & 1023);
  int tp = (int)((idx >> 10) & 127);
  int b  = (int)(idx >> 17);
  int h = d >> 6, hd = d & 63;
  float kv = 0.f, vv = 0.f;
  if (tp < STP) {
    if (tp < STXT) {
      const float* x = enc + ((size_t)b * STXT + tp) * Ctx;
      for (int c = 0; c < Ctx; ++c) {
        float xc = x[c];
        kv += xc * Wk[(size_t)c * D + d];
        vv += xc * Wv[(size_t)c * D + d];
      }
    }
    size_t o = (((size_t)(b * H + h)) * STP + tp) * HD + hd;
    Kt[o] = (_Float16)kv;
    Vt[o] = (_Float16)vv;
  } else {
    int ta = tp - STP;
    if (ta < SAUD) {
      const float* x = aud + ((size_t)b * SAUD + ta) * Ctx;
      for (int c = 0; c < Ctx; ++c) {
        float xc = x[c];
        kv += xc * Wka[(size_t)c * D + d];
        vv += xc * Wva[(size_t)c * D + d];
      }
    }
    size_t o = (((size_t)(b * H + h)) * SAP + ta) * HD + hd;
    Ka[o] = (_Float16)kv;
    Va[o] = (_Float16)vv;
  }
}

// ---------------------------------------------------------------------------
// WMMA GEMM: C[M,N] = A[M,K](f16) * Bt[N,K](f16, pre-transposed) [+ bias]
// Block tile 128x128, 8 waves (4m x 2n), each wave 32x64 = 2x4 WMMA tiles,
// BK=32. Tiles staged into LDS with global_load_async_to_lds_b128.
// ---------------------------------------------------------------------------
template <bool OUT_F16>
__global__ __launch_bounds__(256) void gemm_wmma(
    const _Float16* __restrict__ A, const _Float16* __restrict__ Bt,
    const float* __restrict__ bias, void* __restrict__ Cout,
    int M, int N, int K) {
  __shared__ _Float16 As[128][32];  // [m][k], 8 KB
  __shared__ _Float16 Bs[128][32];  // [n][k], 8 KB

  const int tid  = threadIdx.x;
  const int wave = tid >> 5, lane = tid & 31;
  const int l16  = lane & 15, hi = lane >> 4;
  const int wm = wave & 3, wn = wave >> 2;
  const size_t bm0 = (size_t)blockIdx.y * 128;
  const size_t bn0 = (size_t)blockIdx.x * 128;
  const int m0 = wm * 32, n0 = wn * 64;

  v8f acc[2][4];
#pragma unroll
  for (int i = 0; i < 2; ++i)
#pragma unroll
    for (int j = 0; j < 4; ++j)
#pragma unroll
      for (int r = 0; r < 8; ++r) acc[i][j][r] = 0.0f;

  for (int kb = 0; kb < K; kb += 32) {
    // Stage A (128x32 halves) and B (128x32 halves) via async b128 copies:
    // 512 16-byte chunks each; 2 per thread per matrix.
#pragma unroll
    for (int it = 0; it < 2; ++it) {
      int idx = tid + it * 256;
      int row = idx >> 2;          // 0..127
      int c   = idx & 3;           // 16B chunk within 64B row
      async_copy_b128((uint32_t)(uintptr_t)&As[row][c * 8],
                      A + (bm0 + row) * (size_t)K + kb + c * 8);
      async_copy_b128((uint32_t)(uintptr_t)&Bs[row][c * 8],
                      Bt + (bn0 + row) * (size_t)K + kb + c * 8);
    }
    if (kb + 32 < K)  // hint next A tile toward the caches
      __builtin_prefetch(A + (bm0 + (tid >> 1)) * (size_t)K + kb + 32, 0, 0);
    wait_async0();
    __syncthreads();

    v16h af[2], bf[4];
#pragma unroll
    for (int i = 0; i < 2; ++i) {
      const _Float16* ap = &As[m0 + i * 16 + l16][hi * 16];
#pragma unroll
      for (int j = 0; j < 16; ++j) af[i][j] = ap[j];
    }
#pragma unroll
    for (int i = 0; i < 4; ++i) {
      const _Float16* bp = &Bs[n0 + i * 16 + l16][hi * 16];
#pragma unroll
      for (int j = 0; j < 16; ++j) bf[i][j] = bp[j];
    }
#pragma unroll
    for (int i = 0; i < 2; ++i)
#pragma unroll
      for (int j = 0; j < 4; ++j)
        acc[i][j] = WMMA_F16(af[i], bf[j], acc[i][j]);
    __syncthreads();
  }

#pragma unroll
  for (int i = 0; i < 2; ++i)
#pragma unroll
    for (int j = 0; j < 4; ++j) {
      size_t col = bn0 + n0 + j * 16 + l16;
      float bv = bias ? bias[col] : 0.0f;
#pragma unroll
      for (int r = 0; r < 8; ++r) {
        size_t row = bm0 + m0 + i * 16 + hi * 8 + r;
        float v = acc[i][j][r] + bv;
        if (OUT_F16)
          ((_Float16*)Cout)[row * (size_t)N + col] = (_Float16)v;
        else
          ((float*)Cout)[row * (size_t)N + col] = v;
      }
    }
}

// ---------------------------------------------------------------------------
// Fused dual cross-attention. One wave per (b, h, 16-row q tile).
// Text: 6 score tiles over 96 padded cols (mask >=77); audio: 1 tile (16).
// ---------------------------------------------------------------------------
__global__ __launch_bounds__(256) void attn_kernel(
    const _Float16* __restrict__ Q,
    const _Float16* __restrict__ Kt, const _Float16* __restrict__ Vt,
    const _Float16* __restrict__ Ka, const _Float16* __restrict__ Va,
    _Float16* __restrict__ O) {
  __shared__ _Float16 Pl[8][16][128];  // per-wave probs: [0,96)=text, [96,128)=audio(+pad)

  const int tid  = threadIdx.x;
  const int wave = tid >> 5, lane = tid & 31;
  const int l16  = lane & 15, hi = lane >> 4;
  const int gw = blockIdx.x * 8 + wave;
  const int qt = gw & 255;
  const int h  = (gw >> 8) & 15;
  const int b  = gw >> 12;
  const int q0 = qt * 16;
  const size_t bh = (size_t)(b * H + h);

  // Q A-fragments (two k-steps over HD=64)
  v16h aq[2];
  const _Float16* Qp = Q + ((size_t)(b * S) + q0 + l16) * D + h * HD;
#pragma unroll
  for (int s = 0; s < 2; ++s) {
    const _Float16* qp = Qp + s * 32 + hi * 16;
#pragma unroll
    for (int j = 0; j < 16; ++j) aq[s][j] = qp[j];
  }

  // ---- text scores: 6 n-tiles of 16 tokens ----
  v8f st[6];
#pragma unroll
  for (int t = 0; t < 6; ++t) {
    v8f c;
#pragma unroll
    for (int r = 0; r < 8; ++r) c[r] = 0.0f;
#pragma unroll
    for (int s = 0; s < 2; ++s) {
      v16h bk;
      const _Float16* kp = Kt + (bh * STP + t * 16 + l16) * HD + s * 32 + hi * 16;
#pragma unroll
      for (int j = 0; j < 16; ++j) bk[j] = kp[j];
      c = WMMA_F16(aq[s], bk, c);
    }
    st[t] = c;
  }
  // ---- audio scores: 1 n-tile ----
  v8f sa;
#pragma unroll
  for (int r = 0; r < 8; ++r) sa[r] = 0.0f;
#pragma unroll
  for (int s = 0; s < 2; ++s) {
    v16h bk;
    const _Float16* kp = Ka + (bh * SAP + l16) * HD + s * 32 + hi * 16;
#pragma unroll
    for (int j = 0; j < 16; ++j) bk[j] = kp[j];
    sa = WMMA_F16(aq[s], bk, sa);
  }

  // ---- softmax (per row r; cols spread across 16-lane groups) ----
#pragma unroll
  for (int r = 0; r < 8; ++r) {
    float m = -3.0e38f;
#pragma unroll
    for (int t = 0; t < 6; ++t) {
      int c = t * 16 + l16;
      float sc = (c < STXT) ? st[t][r] * 0.125f : -3.0e38f;
      st[t][r] = sc;
      m = fmaxf(m, sc);
    }
    m = fmaxf(m, swz_xor<1>(m)); m = fmaxf(m, swz_xor<2>(m));
    m = fmaxf(m, swz_xor<4>(m)); m = fmaxf(m, swz_xor<8>(m));
    float sum = 0.f;
#pragma unroll
    for (int t = 0; t < 6; ++t) {
      int c = t * 16 + l16;
      float e = (c < STXT) ? __expf(st[t][r] - m) : 0.f;
      st[t][r] = e;
      sum += e;
    }
    sum += swz_xor<1>(sum); sum += swz_xor<2>(sum);
    sum += swz_xor<4>(sum); sum += swz_xor<8>(sum);
    float inv = 1.0f / sum;
#pragma unroll
    for (int t = 0; t < 6; ++t) st[t][r] *= inv;

    float ma = sa[r] * 0.125f;
    float mm = ma;
    mm = fmaxf(mm, swz_xor<1>(mm)); mm = fmaxf(mm, swz_xor<2>(mm));
    mm = fmaxf(mm, swz_xor<4>(mm)); mm = fmaxf(mm, swz_xor<8>(mm));
    float ea = __expf(ma - mm);
    float sma = ea;
    sma += swz_xor<1>(sma); sma += swz_xor<2>(sma);
    sma += swz_xor<4>(sma); sma += swz_xor<8>(sma);
    sa[r] = ea / sma;
  }

  // ---- redistribute probs via LDS (C-layout -> A-frag layout) ----
#pragma unroll
  for (int t = 0; t < 6; ++t)
#pragma unroll
    for (int r = 0; r < 8; ++r)
      Pl[wave][hi * 8 + r][t * 16 + l16] = (_Float16)st[t][r];
#pragma unroll
  for (int r = 0; r < 8; ++r) {
    Pl[wave][hi * 8 + r][96 + l16]  = (_Float16)sa[r];
    Pl[wave][hi * 8 + r][112 + l16] = (_Float16)0.0f;  // audio pad cols
  }
  __syncthreads();

  // ---- PV: out[16x64] = P_text @ V_text + P_audio @ V_audio ----
  v8f o[4];
#pragma unroll
  for (int n = 0; n < 4; ++n)
#pragma unroll
    for (int r = 0; r < 8; ++r) o[n][r] = 0.0f;

#pragma unroll
  for (int s = 0; s < 3; ++s) {  // text: 96 = 3 k-steps
    v16h ap;
    const _Float16* pp = &Pl[wave][l16][s * 32 + hi * 16];
#pragma unroll
    for (int j = 0; j < 16; ++j) ap[j] = pp[j];
#pragma unroll
    for (int n = 0; n < 4; ++n) {
      v16h bv;
      const _Float16* vp = Vt + (bh * STP + s * 32 + hi * 16) * HD + n * 16 + l16;
#pragma unroll
      for (int j = 0; j < 16; ++j) bv[j] = vp[j * HD];
      o[n] = WMMA_F16(ap, bv, o[n]);
    }
  }
  {  // audio: 1 k-step
    v16h ap;
    const _Float16* pp = &Pl[wave][l16][96 + hi * 16];
#pragma unroll
    for (int j = 0; j < 16; ++j) ap[j] = pp[j];
#pragma unroll
    for (int n = 0; n < 4; ++n) {
      v16h bv;
      const _Float16* vp = Va + (bh * SAP + hi * 16) * HD + n * 16 + l16;
#pragma unroll
      for (int j = 0; j < 16; ++j) bv[j] = vp[j * HD];
      o[n] = WMMA_F16(ap, bv, o[n]);
    }
  }

  // ---- store attention output [b][q][h*64+d] as f16 ----
#pragma unroll
  for (int n = 0; n < 4; ++n) {
    int d = n * 16 + l16;
#pragma unroll
    for (int r = 0; r < 8; ++r) {
      size_t q = (size_t)q0 + hi * 8 + r;
      O[((size_t)(b * S) + q) * D + h * HD + d] = (_Float16)o[n][r];
    }
  }
}

// ---------------------------------------------------------------------------
extern "C" void kernel_launch(void* const* d_in, const int* in_sizes, int n_in,
                              void* d_out, int out_size, void* d_ws, size_t ws_size,
                              hipStream_t stream) {
  const float* hidden = (const float*)d_in[0];
  const float* enc    = (const float*)d_in[1];
  const float* aud    = (const float*)d_in[2];
  const float* Wq     = (const float*)d_in[3];
  const float* Wk     = (const float*)d_in[4];
  const float* Wv     = (const float*)d_in[5];
  const float* Wka    = (const float*)d_in[6];
  const float* Wva    = (const float*)d_in[7];
  const float* Wo     = (const float*)d_in[8];
  const float* bo     = (const float*)d_in[9];
  float* out = (float*)d_out;

  char* ws = (char*)d_ws;
  size_t off = 0;
  auto walloc = [&](size_t bytes) -> void* {
    void* p = ws + off;
    off += (bytes + 255) & ~(size_t)255;
    return p;
  };
  _Float16* Xh   = (_Float16*)walloc((size_t)Bsz * S * D * 2);  // reused as attn out
  _Float16* Qh   = (_Float16*)walloc((size_t)Bsz * S * D * 2);
  _Float16* WqT  = (_Float16*)walloc((size_t)D * D * 2);
  _Float16* WoT  = (_Float16*)walloc((size_t)D * D * 2);
  _Float16* Ktex = (_Float16*)walloc((size_t)Bsz * H * STP * HD * 2);
  _Float16* Vtex = (_Float16*)walloc((size_t)Bsz * H * STP * HD * 2);
  _Float16* Kaud = (_Float16*)walloc((size_t)Bsz * H * SAP * HD * 2);
  _Float16* Vaud = (_Float16*)walloc((size_t)Bsz * H * SAP * HD * 2);

  int n = Bsz * S * D;
  cvt_f32_f16<<<4096, 256, 0, stream>>>(hidden, Xh, n);
  cvt_transpose_f32_f16<<<(D * D) / 256, 256, 0, stream>>>(Wq, WqT, D, D);
  cvt_transpose_f32_f16<<<(D * D) / 256, 256, 0, stream>>>(Wo, WoT, D, D);
  kv_proj_kernel<<<(Bsz * 128 * 1024) / 256, 256, 0, stream>>>(
      enc, aud, Wk, Wv, Wka, Wva, Ktex, Vtex, Kaud, Vaud);
  gemm_wmma<true><<<dim3(D / 128, (Bsz * S) / 128), 256, 0, stream>>>(
      Xh, WqT, (const float*)nullptr, (void*)Qh, Bsz * S, D, D);
  attn_kernel<<<(Bsz * H * (S / 16)) / 8, 256, 0, stream>>>(
      Qh, Ktex, Vtex, Kaud, Vaud, Xh);
  gemm_wmma<false><<<dim3(D / 128, (Bsz * S) / 128), 256, 0, stream>>>(
      Xh, WoT, bo, (void*)out, Bsz * S, D, D);
}
